// RNNPool_14559939133753
// MI455X (gfx1250) — matrix operand order repaired and checked by reading, not compiled, not run on hardware
//
#include <hip/hip_runtime.h>
#include <hip/hip_bf16.h>

typedef __attribute__((ext_vector_type(16))) _Float16 v16h;
typedef __attribute__((ext_vector_type(8)))  float    v8f;

#define CDIM 16
#define RDIM 8
#define CLDIM 8
#define HDIM 32
#define WAVES 4
#define SLOT (16*32)   // one 16x32 f16 tile per slot

__device__ __forceinline__ float fsig(float x){ return 1.0f/(1.0f + __expf(-x)); }
__device__ __forceinline__ float ftanh(float x){ return 2.0f/(1.0f + __expf(-2.0f*x)) - 1.0f; }

// --- A-matrix fragment (16x32 f16) gathered from an LDS tile stored row-major [16][32].
// ISA 7.12.2: lanes 0-15 row m=lane, VGPR0-3 K=0..7, VGPR4-7 K=16..23;
//             lanes 16-31 same rows, K=8..15 and K=24..31.
__device__ __forceinline__ v16h gather_A_lds(const _Float16* buf, int lane){
  const int m  = lane & 15;
  const int kb = (lane >> 4) << 3;
  v16h a;
  #pragma unroll
  for (int i = 0; i < 16; ++i){
    const int k = ((i < 8) ? 0 : 16) + kb + (i & 7);
    a[i] = buf[m*32 + k];
  }
  return a;
}

// --- store C/D-layout f32 accumulators (two 16x16 halves of a 16x32 tile) to LDS as f16.
// C/D layout: lane holds column n=lane&15, rows (0..7)+8*(lane>=16) across the 8 VGPRs.
__device__ __forceinline__ void store_h_lds(_Float16* buf, int lane, const v8f h0, const v8f h1){
  const int n  = lane & 15;
  const int mb = (lane >> 4) << 3;
  #pragma unroll
  for (int j = 0; j < 8; ++j){
    buf[(mb + j)*32 + n]      = (_Float16)h0[j];
    buf[(mb + j)*32 + 16 + n] = (_Float16)h1[j];
  }
}

// --- B-matrix fragment (32x16 f16) for hidden-half `half` from weight matrix W[H][Kd]
// (we need W^T as B: B[k][n] = W[n][k]).  B layout: lanes 0-15 K=0..7,16..23,
// lanes 16-31 K=8..15,24..31, element i -> K = (i<8?0:16)+8*(lane>=16)+(i&7), N = lane&15.
// K entries >= Kd are zero-padded (used for the K=16 input projection).
__device__ __forceinline__ v16h load_B_frag(const float* __restrict__ W, int Kd, int half, int lane){
  const int n  = (lane & 15) + half*16;
  const int kb = (lane >> 4) << 3;
  v16h b;
  #pragma unroll
  for (int i = 0; i < 16; ++i){
    const int k = ((i < 8) ? 0 : 16) + kb + (i & 7);
    b[i] = (k < Kd) ? (_Float16)W[n*Kd + k] : (_Float16)0.0f;
  }
  return b;
}

// --- A-fragment of the input projection operand: rows = 16 batch elems, K = 16 channels
// (zero-padded to K=32).  x layout: [B][C][R][CL] row-major.
__device__ __forceinline__ v16h gather_x_frag(const float* __restrict__ x, int b0,
                                              int rr, int ccl, int lane){
  const int m  = lane & 15;
  const int kb = (lane >> 4) << 3;
  const int base = ((b0 + m) * CDIM) * (RDIM*CLDIM) + rr*CLDIM + ccl;
  v16h a;
  #pragma unroll
  for (int i = 0; i < 16; ++i){
    if (i < 8){
      const int c = kb + i;                       // channels 0..15
      a[i] = (_Float16)x[base + c*(RDIM*CLDIM)];
    } else {
      a[i] = (_Float16)0.0f;                      // K=16..31 zero pad
    }
  }
  return a;
}

// FastGRNN elementwise update on one 16x16 accumulator half (8 values / lane).
__device__ __forceinline__ void grnn_update(v8f& h, const v8f pre, float bg, float bu,
                                            float zf, float nf){
  #pragma unroll
  for (int j = 0; j < 8; ++j){
    float z = fsig(pre[j] + bg);
    float c = ftanh(pre[j] + bu);
    h[j] = z*h[j] + (zf*(1.0f - z) + nf)*c;
  }
}

#define WMMA(A,B,C) __builtin_amdgcn_wmma_f32_16x16x32_f16(false, (A), false, (B), (short)0, (C), false, false)

__global__ __launch_bounds__(WAVES*32, 1)
void rnnpool_kernel(const float* __restrict__ x,
                    const float* __restrict__ W1, const float* __restrict__ U1,
                    const float* __restrict__ bg1, const float* __restrict__ bu1,
                    const float* __restrict__ zeta1, const float* __restrict__ nu1,
                    const float* __restrict__ W2, const float* __restrict__ U2,
                    const float* __restrict__ bg2, const float* __restrict__ bu2,
                    const float* __restrict__ zeta2, const float* __restrict__ nu2,
                    float* __restrict__ out)
{
  // slots 0..7: layer-1 strip results / per-step h transpose buffers
  // slots 8,9 : layer-2 forward / backward h transpose buffers
  __shared__ _Float16 lds[WAVES][10][SLOT];
  const int lane = threadIdx.x & 31;
  const int w    = threadIdx.x >> 5;
  const int b0   = (blockIdx.x * WAVES + w) * 16;
  const bool rowpass = (blockIdx.y == 0);   // rows pass -> out_cols [64:128]; cols pass -> out_rows [0:64]

  const int nlo = lane & 15;

  // ======== layer 1: 8 strips, processed in independent pairs for ILP ========
  {
    const v16h fW1_0 = load_B_frag(W1, CDIM, 0, lane);
    const v16h fW1_1 = load_B_frag(W1, CDIM, 1, lane);
    const v16h fU1_0 = load_B_frag(U1, HDIM, 0, lane);
    const v16h fU1_1 = load_B_frag(U1, HDIM, 1, lane);
    const float zf1 = fsig(zeta1[0]), nf1 = fsig(nu1[0]);
    const float bGa = bg1[nlo], bGb = bg1[16+nlo];
    const float bUa = bu1[nlo], bUb = bu1[16+nlo];

    for (int sp = 0; sp < 4; ++sp){
      const int sA = 2*sp, sB = 2*sp + 1;
      _Float16* bufA = &lds[w][sA][0];
      _Float16* bufB = &lds[w][sB][0];
      v8f hA0 = {}, hA1 = {}, hB0 = {}, hB1 = {};
      store_h_lds(bufA, lane, hA0, hA1);
      store_h_lds(bufB, lane, hB0, hB1);
      for (int t = 0; t < 8; ++t){
        const int rrA  = rowpass ? t  : sA,  cclA = rowpass ? sA : t;
        const int rrB  = rowpass ? t  : sB,  cclB = rowpass ? sB : t;
        const v16h axA = gather_x_frag(x, b0, rrA, cclA, lane);
        const v16h axB = gather_x_frag(x, b0, rrB, cclB, lane);
        v8f zc = {};
        v8f xwA0 = WMMA(axA, fW1_0, zc);
        v8f xwB0 = WMMA(axB, fW1_0, zc);
        v8f xwA1 = WMMA(axA, fW1_1, zc);
        v8f xwB1 = WMMA(axB, fW1_1, zc);
        const v16h ahA = gather_A_lds(bufA, lane);
        const v16h ahB = gather_A_lds(bufB, lane);
        v8f preA0 = WMMA(ahA, fU1_0, xwA0);
        v8f preB0 = WMMA(ahB, fU1_0, xwB0);
        v8f preA1 = WMMA(ahA, fU1_1, xwA1);
        v8f preB1 = WMMA(ahB, fU1_1, xwB1);
        grnn_update(hA0, preA0, bGa, bUa, zf1, nf1);
        grnn_update(hB0, preB0, bGa, bUa, zf1, nf1);
        grnn_update(hA1, preA1, bGb, bUb, zf1, nf1);
        grnn_update(hB1, preB1, bGb, bUb, zf1, nf1);
        store_h_lds(bufA, lane, hA0, hA1);          // final t leaves strip result in slot
        store_h_lds(bufB, lane, hB0, hB1);
      }
    }
  }

  // ======== layer 2: bidirectional FastGRNN, both directions interleaved ========
  {
    const v16h fW2_0 = load_B_frag(W2, HDIM, 0, lane);
    const v16h fW2_1 = load_B_frag(W2, HDIM, 1, lane);
    const v16h fU2_0 = load_B_frag(U2, HDIM, 0, lane);
    const v16h fU2_1 = load_B_frag(U2, HDIM, 1, lane);
    const float zf2 = fsig(zeta2[0]), nf2 = fsig(nu2[0]);
    const float bGa = bg2[nlo], bGb = bg2[16+nlo];
    const float bUa = bu2[nlo], bUb = bu2[16+nlo];

    _Float16* bufF = &lds[w][8][0];
    _Float16* bufR = &lds[w][9][0];
    v8f hF0 = {}, hF1 = {}, hR0 = {}, hR1 = {};
    store_h_lds(bufF, lane, hF0, hF1);
    store_h_lds(bufR, lane, hR0, hR1);
    for (int tt = 0; tt < 8; ++tt){
      const v16h ainF = gather_A_lds(&lds[w][tt][0],     lane);
      const v16h ainR = gather_A_lds(&lds[w][7 - tt][0], lane);
      v8f zc = {};
      v8f xwF0 = WMMA(ainF, fW2_0, zc);
      v8f xwR0 = WMMA(ainR, fW2_0, zc);
      v8f xwF1 = WMMA(ainF, fW2_1, zc);
      v8f xwR1 = WMMA(ainR, fW2_1, zc);
      const v16h ahF = gather_A_lds(bufF, lane);
      const v16h ahR = gather_A_lds(bufR, lane);
      v8f preF0 = WMMA(ahF, fU2_0, xwF0);
      v8f preR0 = WMMA(ahR, fU2_0, xwR0);
      v8f preF1 = WMMA(ahF, fU2_1, xwF1);
      v8f preR1 = WMMA(ahR, fU2_1, xwR1);
      grnn_update(hF0, preF0, bGa, bUa, zf2, nf2);
      grnn_update(hR0, preR0, bGa, bUa, zf2, nf2);
      grnn_update(hF1, preF1, bGb, bUb, zf2, nf2);
      grnn_update(hR1, preR1, bGb, bUb, zf2, nf2);
      store_h_lds(bufF, lane, hF0, hF1);
      store_h_lds(bufR, lane, hR0, hR1);
    }

    // write finals: [outoff+0:32] = forward, [outoff+32:64] = backward
    const int outoff = rowpass ? 64 : 0;
    const int mb = (lane >> 4) << 3;
    #pragma unroll
    for (int j = 0; j < 8; ++j){
      const int rowbase = (b0 + mb + j)*128 + outoff;
      out[rowbase + nlo]           = hF0[j];
      out[rowbase + 16 + nlo]      = hF1[j];
      out[rowbase + 32 + nlo]      = hR0[j];
      out[rowbase + 48 + nlo]      = hR1[j];
    }
  }
}

extern "C" void kernel_launch(void* const* d_in, const int* in_sizes, int n_in,
                              void* d_out, int out_size, void* d_ws, size_t ws_size,
                              hipStream_t stream) {
  const float* x     = (const float*)d_in[0];
  // d_in[1] = batch_size (unused; derived from in_sizes[0])
  const float* W1    = (const float*)d_in[2];
  const float* U1    = (const float*)d_in[3];
  const float* bg1   = (const float*)d_in[4];
  const float* bu1   = (const float*)d_in[5];
  const float* zeta1 = (const float*)d_in[6];
  const float* nu1   = (const float*)d_in[7];
  const float* W2    = (const float*)d_in[8];
  const float* U2    = (const float*)d_in[9];
  const float* bg2   = (const float*)d_in[10];
  const float* bu2   = (const float*)d_in[11];
  const float* zeta2 = (const float*)d_in[12];
  const float* nu2   = (const float*)d_in[13];

  const int Btot = in_sizes[0] / (CDIM * RDIM * CLDIM);   // 32768
  dim3 grid(Btot / (WAVES * 16), 2);                      // (512, 2): y = pass

  rnnpool_kernel<<<grid, WAVES*32, 0, stream>>>(
      x, W1, U1, bg1, bu1, zeta1, nu1, W2, U2, bg2, bu2, zeta2, nu2,
      (float*)d_out);
}
